// HierachicalEncoder_7352984011048
// MI455X (gfx1250) — matrix operand order repaired and checked by reading, not compiled, not run on hardware
//
#include <hip/hip_runtime.h>
#include <math.h>

// Problem constants (match reference)
#define Nn    50000
#define Ee    800000
#define Hh    2
#define Cc    64
#define Ll    3
#define ETOT  (Ee + Nn)        // 850000 edges incl. self loops
#define SLOPE 0.2f

typedef __attribute__((ext_vector_type(16))) _Float16 v16h;
typedef __attribute__((ext_vector_type(8)))  float    v8f;

// edge e -> (src, dst); self loops appended after the E real edges
__device__ __forceinline__ void edge_of(const int* __restrict__ ei, int e, int& s, int& d) {
  if (e < Ee) { s = ei[e]; d = ei[Ee + e]; }
  else        { s = e - Ee; d = e - Ee; }
}

// float atomic max via sign-split int/uint trick (monotone in IEEE bit order)
__device__ __forceinline__ void atomic_max_f32(float* addr, float v) {
  if (v >= 0.f) atomicMax((int*)addr, __float_as_int(v));
  else          atomicMin((unsigned int*)addr, __float_as_uint(v));
}

// ---------------------------------------------------------------------------
// Kernel A: per-edge attention logits via WMMA.
// One wave handles 16 edges: A = leaky(x[dst]+x[src]) as a 16x64 f16 tile,
// B = att^T (64x16, only 2 live columns), D[m][n] = alpha[edge m][head n].
// Two v_wmma_f32_16x16x32_f16 per tile. Then scatter alpha + atomicMax nmax.
// ---------------------------------------------------------------------------
__global__ __launch_bounds__(256) void gat_alpha_wmma(
    const float* __restrict__ x, const int* __restrict__ ei,
    const float* __restrict__ attL,              // [H][C] for this layer
    float* __restrict__ alpha,                   // [ETOT][H]
    float* __restrict__ nmax)                    // [N][H]
{
  __shared__ _Float16 mT[8][16][64];             // 16 KB: leaky(xs+xd) tiles
  __shared__ float    dT[8][16][16];             //  8 KB: WMMA D spill
  __shared__ int      eS[8][16], eD[8][16];      //  1 KB

  const int w    = threadIdx.x >> 5;
  const int lane = threadIdx.x & 31;
  const int e0   = (blockIdx.x * 8 + w) * 16;
  const int n    = lane & 15;                    // N index (head col) / M row
  const bool hi  = lane >= 16;

  if (lane < 16) {
    int e = e0 + lane; if (e >= ETOT) e = ETOT - 1;   // clamp pad rows
    int s, d; edge_of(ei, e, s, d);
    eS[w][lane] = s; eD[w][lane] = d;
  }
  __syncthreads();

  // Build m-tile: row i handled by the whole wave (64 ch = 32 lanes x float2)
  #pragma unroll 4
  for (int i = 0; i < 16; ++i) {
    const int s = eS[w][i], d = eD[w][i];
    const float2 xs = *(const float2*)(x + s * Cc + lane * 2);
    const float2 xd = *(const float2*)(x + d * Cc + lane * 2);
    float m0 = xs.x + xd.x; m0 = (m0 >= 0.f) ? m0 : SLOPE * m0;
    float m1 = xs.y + xd.y; m1 = (m1 >= 0.f) ? m1 : SLOPE * m1;
    mT[w][i][lane * 2]     = (_Float16)m0;
    mT[w][i][lane * 2 + 1] = (_Float16)m1;
  }
  __syncthreads();

  // B fragments (32x16 per step): lane n<H carries att[n][k], else 0.
  // B k order used here (same as A): lanes 0-15 hold K=0..15, lanes 16-31 hold K=16..31.
  v16h b0, b1;
  #pragma unroll
  for (int j = 0; j < 16; ++j) {
    const int k = (hi ? 16 : 0) + j;
    float v0 = 0.f, v1 = 0.f;
    if (n < Hh) { v0 = attL[n * Cc + k]; v1 = attL[n * Cc + 32 + k]; }
    b0[j] = (_Float16)v0; b1[j] = (_Float16)v1;
  }

  // A fragments (16x32 per step): lo lanes K{0..15}, hi lanes K{16..31}
  v16h a0, a1;
  #pragma unroll
  for (int j = 0; j < 16; ++j) {
    const int k = (hi ? 16 : 0) + j;
    a0[j] = mT[w][n][k];
    a1[j] = mT[w][n][32 + k];
  }

  v8f acc = {};
  acc = __builtin_amdgcn_wmma_f32_16x16x32_f16(false, a0, false, b0,
                                               (short)0, acc, false, false);
  acc = __builtin_amdgcn_wmma_f32_16x16x32_f16(false, a1, false, b1,
                                               (short)0, acc, false, false);

  // D layout: VGPR j holds M=j (lanes 0-15) / M=j+8 (lanes 16-31), N=lane&15
  #pragma unroll
  for (int j = 0; j < 8; ++j) dT[w][j + (hi ? 8 : 0)][n] = acc[j];
  __syncthreads();

  // Extract: 32 lanes -> (edge row, head)
  {
    const int r = lane >> 1, h = lane & 1;
    const int e = e0 + r;
    if (e < ETOT) {
      const float v = dT[w][r][h];
      alpha[e * Hh + h] = v;
      atomic_max_f32(&nmax[eD[w][r] * Hh + h], v);
    }
  }
}

// ---------------------------------------------------------------------------
// Kernel B: a = exp(alpha - max[dst]); alpha <- a; nsum[dst] += a
// ---------------------------------------------------------------------------
__global__ void gat_softmax_pass(const int* __restrict__ ei,
                                 float* __restrict__ alpha,
                                 const float* __restrict__ nmax,
                                 float* __restrict__ nsum)
{
  const int t = blockIdx.x * blockDim.x + threadIdx.x;
  if (t >= ETOT * Hh) return;
  const int e = t >> 1, h = t & 1;
  int s, d; edge_of(ei, e, s, d); (void)s;
  const float a = __expf(alpha[t] - nmax[d * Hh + h]);
  alpha[t] = a;
  atomicAdd(&nsum[d * Hh + h], a);
}

// ---------------------------------------------------------------------------
// Kernel C: head-mean folded into a single scalar per edge:
//   w = 0.5*(a0/s0 + a1/s1);  hnext[dst][:] += w * x[src][:]
// 16 lanes per edge, float4 per lane.
// ---------------------------------------------------------------------------
__global__ __launch_bounds__(256) void gat_scatter(
    const float* __restrict__ xin, const int* __restrict__ ei,
    const float* __restrict__ alpha, const float* __restrict__ nsum,
    float* __restrict__ hnext)
{
  const int e  = blockIdx.x * 16 + (threadIdx.x >> 4);
  const int c4 = (threadIdx.x & 15) * 4;
  if (e >= ETOT) return;
  int s, d; edge_of(ei, e, s, d);
  const float w = 0.5f * (alpha[e * 2]     / (nsum[d * 2]     + 1e-16f)
                        + alpha[e * 2 + 1] / (nsum[d * 2 + 1] + 1e-16f));
  const float4 xv = *(const float4*)(xin + s * Cc + c4);
  float* o = hnext + d * Cc + c4;
  atomicAdd(o + 0, xv.x * w);
  atomicAdd(o + 1, xv.y * w);
  atomicAdd(o + 2, xv.z * w);
  atomicAdd(o + 3, xv.w * w);
}

// ---------------------------------------------------------------------------
// Small elementwise kernels
// ---------------------------------------------------------------------------
__global__ void gat_init_layer(float* __restrict__ hnext,
                               float* __restrict__ nmax,
                               float* __restrict__ nsum)
{
  const int t = blockIdx.x * blockDim.x + threadIdx.x;
  if (t < Nn * Cc) hnext[t] = 0.f;
  if (t < Nn * Hh) { nmax[t] = -INFINITY; nsum[t] = 0.f; }
}

__global__ void gat_init_total(const float* __restrict__ x, float* __restrict__ total)
{
  const int t = blockIdx.x * blockDim.x + threadIdx.x;
  if (t < Nn * Cc) total[t] = x[t];
}

__global__ void gat_finish_layer(float* __restrict__ hnext,
                                 const float* __restrict__ biasL,
                                 float* __restrict__ total)
{
  const int t = blockIdx.x * blockDim.x + threadIdx.x;
  if (t >= Nn * Cc) return;
  const float v = hnext[t] + biasL[t & (Cc - 1)];
  hnext[t] = v;
  total[t] += v;
}

__global__ void gat_final_avg(const float* __restrict__ total, float* __restrict__ out)
{
  const int t = blockIdx.x * blockDim.x + threadIdx.x;
  if (t < Nn * Cc) out[t] = total[t] * 0.25f;   // mean over {x, h1, h2, h3}
}

// ---------------------------------------------------------------------------
extern "C" void kernel_launch(void* const* d_in, const int* in_sizes, int n_in,
                              void* d_out, int out_size, void* d_ws, size_t ws_size,
                              hipStream_t stream) {
  const float* x    = (const float*)d_in[0];   // [N, C]
  const int*   ei   = (const int*)  d_in[1];   // [2, E]
  const float* att  = (const float*)d_in[2];   // [L, H, C]
  const float* bias = (const float*)d_in[3];   // [L, C]
  float*       out  = (float*)d_out;           // [N, C]

  // workspace layout (floats): ~46 MB total
  float* ws    = (float*)d_ws;
  float* total = ws;                           // N*C
  float* h0    = total + (size_t)Nn * Cc;      // N*C
  float* h1    = h0    + (size_t)Nn * Cc;      // N*C
  float* alpha = h1    + (size_t)Nn * Cc;      // ETOT*H
  float* nmax  = alpha + (size_t)ETOT * Hh;    // N*H
  float* nsum  = nmax  + (size_t)Nn * Hh;      // N*H

  const int grid_nc    = (Nn * Cc + 255) / 256;       // node*channel grids
  const int grid_tiles = (ETOT + 127) / 128;          // 8 waves x 16 edges
  const int grid_eh    = (ETOT * Hh + 255) / 256;
  const int grid_scat  = (ETOT + 15) / 16;            // 16 edges / block

  gat_init_total<<<grid_nc, 256, 0, stream>>>(x, total);

  const float* cur = x;
  float* nxt = h0;
  for (int l = 0; l < Ll; ++l) {
    gat_init_layer<<<grid_nc, 256, 0, stream>>>(nxt, nmax, nsum);
    gat_alpha_wmma<<<grid_tiles, 256, 0, stream>>>(cur, ei, att + (size_t)l * Hh * Cc,
                                                   alpha, nmax);
    gat_softmax_pass<<<grid_eh, 256, 0, stream>>>(ei, alpha, nmax, nsum);
    gat_scatter<<<grid_scat, 256, 0, stream>>>(cur, ei, alpha, nsum, nxt);
    gat_finish_layer<<<grid_nc, 256, 0, stream>>>(nxt, bias + (size_t)l * Cc, total);
    cur = nxt;
    nxt = (nxt == h0) ? h1 : h0;
  }

  gat_final_avg<<<grid_nc, 256, 0, stream>>>(total, out);
}
